// LinformerAttentionExample_14087492731296
// MI455X (gfx1250) — matrix operand order repaired and checked by reading, not compiled, not run on hardware
//
#include <hip/hip_runtime.h>

#define DEPTH 6
#define DIM   1024
#define HEADS 16
#define SEQ   2048
#define LRK   256
#define BATCH 4
#define DH    64
#define FFD   4096
#define ROWS  (BATCH*SEQ)   /* 8192 */

typedef __bf16 bf16;
typedef __bf16 v16bf __attribute__((ext_vector_type(16)));
typedef __bf16 v8bf  __attribute__((ext_vector_type(8)));
typedef float  v8f   __attribute__((ext_vector_type(8)));
typedef unsigned int u32x4 __attribute__((ext_vector_type(4)));
typedef int          i32x4 __attribute__((ext_vector_type(4)));
typedef int          i32x8 __attribute__((ext_vector_type(8)));

// Build a 16-element bf16 fragment from two 16-byte (ds/global)_load_b128 chunks.
__device__ __forceinline__ v16bf ld2x8(const bf16* p0, const bf16* p1) {
  v8bf lo = *(const v8bf*)p0;
  v8bf hi = *(const v8bf*)p1;
  v16bf r;
#pragma unroll
  for (int i = 0; i < 8; ++i) { r[i] = lo[i]; r[i + 8] = hi[i]; }
  return r;
}

__device__ __forceinline__ v8f wmma_bf16(v16bf a, v16bf b, v8f c) {
  return __builtin_amdgcn_wmma_f32_16x16x32_bf16(false, a, false, b, (short)0, c,
                                                 false, false);
}

// Wave-relative LDS byte offset of a __shared__ object: flat shared pointers
// carry the LDS offset in their low 32 bits (LDS aperture mapping).
__device__ __forceinline__ unsigned lds_off_of(const void* p) {
  return (unsigned)(unsigned long long)(uintptr_t)p;
}

// ---------------------------------------------------------------------------
// TDM: DMA a 2D bf16 tile (tile_w x tile_h, row stride = stride_elems) from
// global memory into LDS, padding each 64B LDS row by 32B (=> 48-elem rows,
// matching the padded shared-tile layout used by the WMMA fragment loads).
// Tracked with TENSORcnt.  D# layout per CDNA5 ISA 8.3/8.4 (2D, groups 2/3=0).
// ---------------------------------------------------------------------------
__device__ __forceinline__ void tdm_load_tile_2d(unsigned lds_addr,
                                                 unsigned long long gaddr,
                                                 unsigned tile_w, unsigned tile_h,
                                                 unsigned stride_elems) {
  u32x4 g0;
  g0[0] = 1u;                                           // count=1, user mode
  g0[1] = lds_addr;                                     // LDS byte address
  g0[2] = (unsigned)gaddr;                              // global addr [31:0]
  g0[3] = (unsigned)((gaddr >> 32) & 0x01FFFFFFull)     // global addr [56:32]
          | (2u << 30);                                 // type = 2 (image)
  const unsigned TD = 1u << 20;                         // generous tensor dims
  i32x8 g1;
  g1[0] = (int)((1u << 16)      // data_size = 2 bytes
              | (1u << 20)      // pad_enable
              | (3u << 22)      // pad_interval: 16 DWORDs (64B)
              | (7u << 25));    // pad_amount:   8 DWORDs (32B)
  g1[1] = (int)((TD & 0xffffu) << 16);                  // tensor_dim0 [15:0]
  g1[2] = (int)((TD >> 16) | ((TD & 0xffffu) << 16));   // dim0 hi | dim1 lo
  g1[3] = (int)((TD >> 16) | (tile_w << 16));           // dim1 hi | tile_dim0
  g1[4] = (int)(tile_h & 0xffffu);                      // tile_dim1 (tile_dim2=0)
  g1[5] = (int)stride_elems;                            // tensor_dim0_stride lo32
  g1[6] = 0;                                            // stride hi | dim1_stride lo
  g1[7] = 0;
  i32x4 g2 = {};
  i32x4 g3 = {};
#if defined(__clang_major__) && (__clang_major__ >= 23)
  i32x8 g4 = {};
  __builtin_amdgcn_tensor_load_to_lds(g0, g1, g2, g3, g4, 0);
#else
  __builtin_amdgcn_tensor_load_to_lds(g0, g1, g2, g3, 0);
#endif
}

// ---------------------------------------------------------------------------
// Generic bf16 GEMM: C[M,N] = A[M,K] * B[K,N]  (row-major, M%128==0, N%128==0,
// K%32==0).  A tiles are DMA'd by the Tensor Data Mover into a double-buffered
// padded LDS tile (prefetched across the compute); B tiles are staged manually
// (transposed to [n][k] for the WMMA B-fragment layout).
// EPI 0: store bf16.  EPI 1: +bias +fp32 residual -> fp32.  EPI 2: +bias,
// exact GELU -> bf16.  blockIdx.z batches via strides.
// ---------------------------------------------------------------------------
template <int EPI>
__global__ __launch_bounds__(256) void gemm_bf16_kernel(
    const bf16* __restrict__ A, int lda, long long sA,
    const bf16* __restrict__ B, long long sB,
    const float* __restrict__ bias, const float* __restrict__ resid,
    float* __restrict__ Cf, bf16* __restrict__ Cb, long long sC,
    int M, int N, int Kd) {
  (void)M;
  __shared__ __align__(16) bf16 shA[2][128 * 48];  // [m][k] rows padded to 48
  __shared__ __align__(16) bf16 shB[128 * 48];     // [n][k] (transposed)

  const int tid  = threadIdx.x;
  const int wave = tid >> 5;
  const int lane = tid & 31;
  const int wm   = wave >> 1;      // 0..3  -> 32-row band
  const int wn   = wave & 1;       // 0..1  -> 64-col band
  const int ml   = lane & 15;
  const int hi   = lane >> 4;      // lane half
  const int mhalf  = hi ? 8 : 0;   // C-layout row offset
  const int kloA   = hi ? 8 : 0;   // A fragment K offset
  const int khalfB = hi ? 8 : 0;   // B fragment K offset

  const int bm = blockIdx.y * 128;
  const int bn = blockIdx.x * 128;
  A += (long long)blockIdx.z * sA;
  B += (long long)blockIdx.z * sB;
  const long long cbase = (long long)blockIdx.z * sC;

  const int brow = tid >> 3, bcol = (tid & 7) * 16;  // B tile: 32x128

  const unsigned ldsA[2] = {lds_off_of(&shA[0][0]), lds_off_of(&shA[1][0])};
  const unsigned long long abase =
      (unsigned long long)(uintptr_t)A + 2ull * ((long long)bm * lda);

  v8f acc[2][4] = {};

  const int NK = Kd >> 5;
  if (wave == 0) tdm_load_tile_2d(ldsA[0], abase, 32, 128, (unsigned)lda);

  for (int it = 0; it < NK; ++it) {
    const int cur = it & 1;
    // --- stage B tile transposed: shB[n][k] ---
    const bf16* gb = B + (long long)(it * 32 + brow) * N + bn + bcol;
    v8bf b0 = *(const v8bf*)gb;
    v8bf b1 = *(const v8bf*)(gb + 8);
    if (it + 1 < NK)  // pull next B tile toward the caches
      __builtin_prefetch(B + (long long)((it + 1) * 32 + brow) * N + bn + bcol, 0, 0);
#pragma unroll
    for (int i = 0; i < 8; ++i) {
      shB[(bcol + i) * 48 + brow]     = b0[i];
      shB[(bcol + 8 + i) * 48 + brow] = b1[i];
    }
    if (wave == 0) __builtin_amdgcn_s_wait_tensorcnt(0);  // A[cur] landed
    __syncthreads();
    // prefetch next A tile via TDM, overlapping the WMMAs below
    if (wave == 0 && (it + 1 < NK))
      tdm_load_tile_2d(ldsA[cur ^ 1], abase + 2ull * (unsigned long long)((it + 1) * 32),
                       32, 128, (unsigned)lda);

    v16bf af[2], bfr[4];
#pragma unroll
    for (int mt = 0; mt < 2; ++mt) {
      const bf16* pa = &shA[cur][(wm * 32 + mt * 16 + ml) * 48 + kloA];
      af[mt] = ld2x8(pa, pa + 16);        // K: klo..klo+7, klo+16..klo+23
    }
#pragma unroll
    for (int nt = 0; nt < 4; ++nt) {
      const bf16* pb = &shB[(wn * 64 + nt * 16 + ml) * 48 + khalfB];
      bfr[nt] = ld2x8(pb, pb + 16);       // K: klo..klo+7, klo+16..klo+23
    }
#pragma unroll
    for (int mt = 0; mt < 2; ++mt)
#pragma unroll
      for (int nt = 0; nt < 4; ++nt)
        acc[mt][nt] = wmma_bf16(af[mt], bfr[nt], acc[mt][nt]);
    __syncthreads();
  }

  // epilogue (C layout: vgpr r -> row mhalf+r, lane&15 -> col)
#pragma unroll
  for (int mt = 0; mt < 2; ++mt)
#pragma unroll
    for (int nt = 0; nt < 4; ++nt)
#pragma unroll
      for (int r = 0; r < 8; ++r) {
        const int row = bm + wm * 32 + mt * 16 + mhalf + r;
        const int col = bn + wn * 64 + nt * 16 + ml;
        const long long idx = cbase + (long long)row * N + col;
        float v = acc[mt][nt][r];
        if (EPI == 0) {
          Cb[idx] = (bf16)v;
        } else if (EPI == 1) {
          Cf[idx] = v + bias[col] + resid[(long long)row * N + col];
        } else {
          float t = v + bias[col];
          t = 0.5f * t * (1.0f + erff(t * 0.70710678118654752f));
          Cb[idx] = (bf16)t;
        }
      }
}

// ---------------------------------------------------------------------------
// Fused Linformer attention per (batch, head): scores = q*kk^T/8, softmax over
// 256, o = attn * vv.  4 waves x 16 query rows per block.
// ---------------------------------------------------------------------------
__global__ __launch_bounds__(128) void attn_kernel(
    const bf16* __restrict__ q,   // [ROWS, DIM]
    const bf16* __restrict__ kk,  // [BATCH*LRK, DIM]
    const bf16* __restrict__ vv,  // [BATCH*LRK, DIM]
    bf16* __restrict__ o) {       // [ROWS, DIM]
  const int bh = blockIdx.y;
  const int b = bh >> 4, h = bh & 15;
  const int tid = threadIdx.x;
  const int wave = tid >> 5, lane = tid & 31;
  const int ml = lane & 15, hi = lane >> 4;

  __shared__ __align__(16) bf16 vvT[64 * 264];       // [d][k], padded
  __shared__ __align__(16) bf16 attnS[4][16 * 264];  // per-wave [row][k]

  // stage V transposed: vvT[d][k] = vv[b,k,h*64+d]
#pragma unroll 4
  for (int it = 0; it < 16; ++it) {
    int g = it * 128 + tid;           // 2048 groups of 8 d-values
    int k = g >> 3, d0 = (g & 7) * 8;
    v8bf vval = *(const v8bf*)&vv[(long long)(b * LRK + k) * DIM + h * 64 + d0];
#pragma unroll
    for (int i = 0; i < 8; ++i) vvT[(d0 + i) * 264 + k] = vval[i];
  }
  __syncthreads();

  const int rowq = blockIdx.x * 64 + wave * 16;

  // Q A-fragments (held across all 16 key tiles)
  v16bf aq[2];
  {
    const bf16* gq = q + (long long)(b * SEQ + rowq + ml) * DIM + h * 64;
    const int klo = hi ? 8 : 0;
    aq[0] = ld2x8(gq + klo,      gq + klo + 16);
    aq[1] = ld2x8(gq + 32 + klo, gq + 32 + klo + 16);
  }

  // scores: 16 rows x 256 keys in 16 accumulators
  v8f s[16];
#pragma unroll
  for (int kt = 0; kt < 16; ++kt) {
    v8f c = {};
#pragma unroll
    for (int ks = 0; ks < 2; ++ks) {
      const bf16* gk = kk + (long long)(b * LRK + kt * 16 + ml) * DIM +
                       h * 64 + ks * 32 + (hi ? 8 : 0);
      v16bf bfr = ld2x8(gk, gk + 16);
      c = wmma_bf16(aq[ks], bfr, c);
    }
    s[kt] = c;
  }

  // row softmax (scale 1/sqrt(64)=0.125) using half-wave shuffles
#pragma unroll
  for (int r = 0; r < 8; ++r) {
    float mx = -3.0e38f;
#pragma unroll
    for (int kt = 0; kt < 16; ++kt) mx = fmaxf(mx, s[kt][r]);
    mx = fmaxf(mx, __shfl_xor(mx, 1));
    mx = fmaxf(mx, __shfl_xor(mx, 2));
    mx = fmaxf(mx, __shfl_xor(mx, 4));
    mx = fmaxf(mx, __shfl_xor(mx, 8));
    float sum = 0.f;
#pragma unroll
    for (int kt = 0; kt < 16; ++kt) {
      float e = expf((s[kt][r] - mx) * 0.125f);
      s[kt][r] = e;
      sum += e;
    }
    sum += __shfl_xor(sum, 1);
    sum += __shfl_xor(sum, 2);
    sum += __shfl_xor(sum, 4);
    sum += __shfl_xor(sum, 8);
    const float inv = 1.0f / sum;
    const int rr = hi * 8 + r;
#pragma unroll
    for (int kt = 0; kt < 16; ++kt)
      attnS[wave][rr * 264 + kt * 16 + ml] = (bf16)(s[kt][r] * inv);
  }

  // o = attn(16x256) * vv(256x64)
  v8f oacc[4] = {};
  const int kloA = hi ? 8 : 0;
  const int khB  = hi ? 8 : 0;
#pragma unroll
  for (int k2 = 0; k2 < 8; ++k2) {
    const bf16* pa = &attnS[wave][ml * 264 + k2 * 32 + kloA];
    v16bf afr = ld2x8(pa, pa + 16);
#pragma unroll
    for (int dt = 0; dt < 4; ++dt) {
      const bf16* pb = &vvT[(dt * 16 + ml) * 264 + k2 * 32 + khB];
      v16bf bfr = ld2x8(pb, pb + 16);
      oacc[dt] = wmma_bf16(afr, bfr, oacc[dt]);
    }
  }
#pragma unroll
  for (int dt = 0; dt < 4; ++dt)
#pragma unroll
    for (int r = 0; r < 8; ++r) {
      const int row = b * SEQ + rowq + hi * 8 + r;
      const int col = h * 64 + dt * 16 + ml;
      o[(long long)row * DIM + col] = (bf16)oacc[dt][r];
    }
}

// ---------------------------------------------------------------------------
// LayerNorm over DIM=1024, fp32 in -> bf16 out.  One block per row.
// ---------------------------------------------------------------------------
__global__ __launch_bounds__(256) void layernorm_kernel(
    const float* __restrict__ x, const float* __restrict__ g,
    const float* __restrict__ be, bf16* __restrict__ out) {
  const int row = blockIdx.x;
  const int tid = threadIdx.x;
  const int lane = tid & 31, wave = tid >> 5;
  const float* xr = x + (long long)row * DIM;
  float v[4], s1 = 0.f, s2 = 0.f;
#pragma unroll
  for (int i = 0; i < 4; ++i) {
    float t = xr[tid + i * 256];
    v[i] = t;
    s1 += t;
    s2 += t * t;
  }
#pragma unroll
  for (int m = 16; m >= 1; m >>= 1) {
    s1 += __shfl_xor(s1, m);
    s2 += __shfl_xor(s2, m);
  }
  __shared__ float r1[8], r2[8];
  if (lane == 0) { r1[wave] = s1; r2[wave] = s2; }
  __syncthreads();
  if (tid == 0) {
    float a = 0.f, c = 0.f;
    for (int j = 0; j < 8; ++j) { a += r1[j]; c += r2[j]; }
    r1[0] = a;
    r2[0] = c;
  }
  __syncthreads();
  const float mu  = r1[0] * (1.0f / DIM);
  const float var = r2[0] * (1.0f / DIM) - mu * mu;
  const float inv = rsqrtf(var + 1e-5f);
#pragma unroll
  for (int i = 0; i < 4; ++i) {
    const int c = tid + i * 256;
    out[(long long)row * DIM + c] = (bf16)((v[i] - mu) * inv * g[c] + be[c]);
  }
}

// ---------------------------------------------------------------------------
__global__ void cvt_f32_bf16(const float* __restrict__ in, bf16* __restrict__ out,
                             int n) {
  int i = blockIdx.x * blockDim.x + threadIdx.x;
  const int st = gridDim.x * blockDim.x;
  for (; i < n; i += st) out[i] = (bf16)in[i];
}

// proj [SEQ][LRK] fp32 -> [LRK][SEQ] bf16
__global__ void transpose_cvt(const float* __restrict__ in, bf16* __restrict__ out) {
  const int i = blockIdx.x * 256 + threadIdx.x;  // over LRK*SEQ
  const int k = i >> 11, n = i & 2047;
  out[i] = (bf16)in[n * LRK + k];
}

__global__ void copy_f32(const float* __restrict__ in, float* __restrict__ out,
                         int n) {
  int i = blockIdx.x * blockDim.x + threadIdx.x;
  const int st = gridDim.x * blockDim.x;
  for (; i < n; i += st) out[i] = in[i];
}

// ---------------------------------------------------------------------------
extern "C" void kernel_launch(void* const* d_in, const int* in_sizes, int n_in,
                              void* d_out, int out_size, void* d_ws, size_t ws_size,
                              hipStream_t stream) {
  (void)in_sizes; (void)n_in; (void)out_size; (void)ws_size;
  const float* x    = (const float*)d_in[0];
  const float* ln1g = (const float*)d_in[1];
  const float* ln1b = (const float*)d_in[2];
  const float* Wq   = (const float*)d_in[3];
  const float* Wk   = (const float*)d_in[4];
  const float* Wv   = (const float*)d_in[5];
  const float* Pk   = (const float*)d_in[6];
  const float* Pv   = (const float*)d_in[7];
  const float* Wo   = (const float*)d_in[8];
  const float* bo   = (const float*)d_in[9];
  const float* ln2g = (const float*)d_in[10];
  const float* ln2b = (const float*)d_in[11];
  const float* W1   = (const float*)d_in[12];
  const float* b1   = (const float*)d_in[13];
  const float* W2   = (const float*)d_in[14];
  const float* b2   = (const float*)d_in[15];

  char* ws = (char*)d_ws;
  size_t off = 0;
  auto take = [&](size_t bytes) -> char* {
    off = (off + 255) & ~(size_t)255;
    char* p = ws + off;
    off += bytes;
    return p;
  };
  bf16* wq_b  = (bf16*)take((size_t)DIM * DIM * 2);
  bf16* wk_b  = (bf16*)take((size_t)DIM * DIM * 2);
  bf16* wv_b  = (bf16*)take((size_t)DIM * DIM * 2);
  bf16* wo_b  = (bf16*)take((size_t)DIM * DIM * 2);
  bf16* w1_b  = (bf16*)take((size_t)DIM * FFD * 2);
  bf16* w2_b  = (bf16*)take((size_t)FFD * DIM * 2);
  bf16* pkT_b = (bf16*)take((size_t)LRK * SEQ * 2);
  bf16* pvT_b = (bf16*)take((size_t)LRK * SEQ * 2);
  float* xcur = (float*)take((size_t)ROWS * DIM * 4);
  bf16* hb    = (bf16*)take((size_t)ROWS * DIM * 2);
  bf16* qb    = (bf16*)take((size_t)ROWS * DIM * 2);
  bf16* hkb   = (bf16*)take((size_t)ROWS * DIM * 2);
  bf16* hvb   = (bf16*)take((size_t)ROWS * DIM * 2);
  bf16* kkb   = (bf16*)take((size_t)BATCH * LRK * DIM * 2);
  bf16* vvb   = (bf16*)take((size_t)BATCH * LRK * DIM * 2);
  bf16* ob    = (bf16*)take((size_t)ROWS * DIM * 2);
  bf16* ffb   = (bf16*)take((size_t)ROWS * FFD * 2);

  copy_f32<<<4096, 256, 0, stream>>>(x, xcur, ROWS * DIM);

  for (int l = 0; l < DEPTH; ++l) {
    cvt_f32_bf16<<<4096, 256, 0, stream>>>(Wq + (size_t)l * DIM * DIM, wq_b, DIM * DIM);
    cvt_f32_bf16<<<4096, 256, 0, stream>>>(Wk + (size_t)l * DIM * DIM, wk_b, DIM * DIM);
    cvt_f32_bf16<<<4096, 256, 0, stream>>>(Wv + (size_t)l * DIM * DIM, wv_b, DIM * DIM);
    cvt_f32_bf16<<<4096, 256, 0, stream>>>(Wo + (size_t)l * DIM * DIM, wo_b, DIM * DIM);
    cvt_f32_bf16<<<4096, 256, 0, stream>>>(W1 + (size_t)l * DIM * FFD, w1_b, DIM * FFD);
    cvt_f32_bf16<<<4096, 256, 0, stream>>>(W2 + (size_t)l * FFD * DIM, w2_b, FFD * DIM);
    transpose_cvt<<<(LRK * SEQ) / 256, 256, 0, stream>>>(Pk + (size_t)l * SEQ * LRK, pkT_b);
    transpose_cvt<<<(LRK * SEQ) / 256, 256, 0, stream>>>(Pv + (size_t)l * SEQ * LRK, pvT_b);

    // PreNorm 1
    layernorm_kernel<<<ROWS, 256, 0, stream>>>(xcur, ln1g + l * DIM, ln1b + l * DIM, hb);

    // q / k / v projections (8192 x 1024 x 1024)
    gemm_bf16_kernel<0><<<dim3(DIM / 128, ROWS / 128, 1), 256, 0, stream>>>(
        hb, DIM, 0, wq_b, 0, nullptr, nullptr, nullptr, qb, 0, ROWS, DIM, DIM);
    gemm_bf16_kernel<0><<<dim3(DIM / 128, ROWS / 128, 1), 256, 0, stream>>>(
        hb, DIM, 0, wk_b, 0, nullptr, nullptr, nullptr, hkb, 0, ROWS, DIM, DIM);
    gemm_bf16_kernel<0><<<dim3(DIM / 128, ROWS / 128, 1), 256, 0, stream>>>(
        hb, DIM, 0, wv_b, 0, nullptr, nullptr, nullptr, hvb, 0, ROWS, DIM, DIM);

    // low-rank K/V: kk[b] = Pk^T (256x2048) @ hk[b] (2048x1024), batched z=4
    gemm_bf16_kernel<0><<<dim3(DIM / 128, LRK / 128, BATCH), 256, 0, stream>>>(
        pkT_b, SEQ, 0, hkb, (long long)SEQ * DIM, nullptr, nullptr, nullptr, kkb,
        (long long)LRK * DIM, LRK, DIM, SEQ);
    gemm_bf16_kernel<0><<<dim3(DIM / 128, LRK / 128, BATCH), 256, 0, stream>>>(
        pvT_b, SEQ, 0, hvb, (long long)SEQ * DIM, nullptr, nullptr, nullptr, vvb,
        (long long)LRK * DIM, LRK, DIM, SEQ);

    // fused attention
    attn_kernel<<<dim3(SEQ / 64, BATCH * HEADS), 128, 0, stream>>>(qb, kkb, vvb, ob);

    // x += o @ Wo + bo
    gemm_bf16_kernel<1><<<dim3(DIM / 128, ROWS / 128, 1), 256, 0, stream>>>(
        ob, DIM, 0, wo_b, 0, bo + l * DIM, xcur, xcur, nullptr, 0, ROWS, DIM, DIM);

    // PreNorm 2 + FFN
    layernorm_kernel<<<ROWS, 256, 0, stream>>>(xcur, ln2g + l * DIM, ln2b + l * DIM, hb);
    gemm_bf16_kernel<2><<<dim3(FFD / 128, ROWS / 128, 1), 256, 0, stream>>>(
        hb, DIM, 0, w1_b, 0, b1 + (size_t)l * FFD, nullptr, nullptr, ffb, 0,
        ROWS, FFD, DIM);
    float* outp = (l == DEPTH - 1) ? (float*)d_out : xcur;
    gemm_bf16_kernel<1><<<dim3(DIM / 128, ROWS / 128, 1), 256, 0, stream>>>(
        ffb, FFD, 0, w2_b, 0, b2 + l * DIM, xcur, outp, nullptr, 0, ROWS, DIM, FFD);
  }
}
